// Zoner_11940009083534
// MI455X (gfx1250) — compile-verified, matrix operand order checked by reading, not run on hardware
//
#include <hip/hip_runtime.h>
#include <hip/hip_bf16.h>
#include <math.h>

// Problem dims (compile-time constants from the reference)
#define B_  32
#define L_  80
#define Z_  4096
#define D_  768
#define DO_ 256
#define KT_ (D_ / 32)      // 24 K-steps of 32
#define NT_ (DO_ / 16)     // 16 N-tiles of 16
#define LDS_STRIDE 776     // 768 + 8 bf16 pad (16B) to stagger banks
#define INV_SQRT_FD 0.03608439182435161f  // 1/sqrt(768)

typedef __attribute__((ext_vector_type(16))) __bf16          v16bf;
typedef __attribute__((ext_vector_type(8)))  float           v8f;
typedef __attribute__((ext_vector_type(8)))  unsigned short  ushort8;
typedef __attribute__((ext_vector_type(4)))  unsigned short  ushort4_t;

union FragBF { v16bf v; ushort8 h[2]; };

__device__ __forceinline__ unsigned short f2bf(float x) {
  unsigned int u = __float_as_uint(x);
  unsigned int r = u + 0x7FFFu + ((u >> 16) & 1u);   // round-to-nearest-even
  return (unsigned short)(r >> 16);
}

// Branchless tanh: 1 - 2/(exp(2x)+1). Saturates correctly at +/-inf, monotone.
// Uses raw v_exp_f32 + v_rcp_f32 (TRANS ops) -- no IEEE div sequence, no
// EXEC-mask divergence (libm tanhf lowers to ~40 branchy VALU).
__device__ __forceinline__ float fast_tanh(float x) {
  const float e = __expf(2.0f * x) + 1.0f;
  return 1.0f - 2.0f * __builtin_amdgcn_rcpf(e);
}

// ---------------------------------------------------------------------------
// Kernel 1: t = tanh(mean_L(txt) @ Wt + bt)   [B, DO]
// ---------------------------------------------------------------------------
__global__ __launch_bounds__(256) void prep_t_kernel(
    const float* __restrict__ txt, const float* __restrict__ Wt,
    const float* __restrict__ bt, float* __restrict__ tout)
{
  __shared__ float meanv[D_];
  const int b = blockIdx.x, tid = threadIdx.x;
  for (int d = tid; d < D_; d += 256) {
    float s = 0.0f;
    const float* p = txt + (size_t)b * L_ * D_ + d;
    for (int l = 0; l < L_; ++l) s += p[(size_t)l * D_];
    meanv[d] = s * (1.0f / (float)L_);
  }
  __syncthreads();
  float acc = bt[tid];
  for (int d = 0; d < D_; ++d) acc += meanv[d] * Wt[(size_t)d * DO_ + tid];
  tout[(size_t)b * DO_ + tid] = fast_tanh(acc);
}

// ---------------------------------------------------------------------------
// Kernel 2: pack Wz (fp32 [D, DO]) -> bf16 WMMA B-fragments.
// Block index (kt, nt); within block: 32 lanes x 16 bf16, contiguous per lane.
// Lane/K mapping per CDNA5 ISA 7.12.2 (16-bit matrices):
//   lane<16 : e<8 -> K=e      ; e>=8 -> K=e+8
//   lane>=16: e<8 -> K=e+8    ; e>=8 -> K=e+16
//   N = nt*16 + (lane & 15)
// ---------------------------------------------------------------------------
__global__ __launch_bounds__(256) void pack_wz_kernel(
    const float* __restrict__ Wz, unsigned short* __restrict__ out)
{
  const int idx  = blockIdx.x * 256 + threadIdx.x;   // < KT_*NT_*32*16 = 196608
  const int e    = idx & 15;
  const int lane = (idx >> 4) & 31;
  const int nt   = (idx >> 9) & 15;
  const int kt   = idx >> 13;
  int K;
  if (lane < 16) K = (e < 8) ? e       : (e + 8);
  else           K = (e < 8) ? (e + 8) : (e + 16);
  const int n = nt * 16 + (lane & 15);
  out[idx] = f2bf(Wz[(size_t)(kt * 32 + K) * DO_ + n]);
}

// ---------------------------------------------------------------------------
// Kernel 3 (main, fused): per (b, 16-row zone tile):
//   attn[16] = sum_n tanh( (zone16x768 @ Wz)[m,n] + bz[n] ) * t[b,n] / sqrt(768)
// 512 threads = 16 waves; wave w owns output columns [16w, 16w+16).
// ---------------------------------------------------------------------------
__global__ __launch_bounds__(512) void zone_attn_kernel(
    const float* __restrict__ zone, const unsigned short* __restrict__ wz_bf,
    const float* __restrict__ tvec, const float* __restrict__ bz,
    float* __restrict__ attn)
{
  const int ztile = blockIdx.x;      // 0..255
  const int b     = blockIdx.y;      // 0..31
  const int tid   = threadIdx.x;     // 0..511
  const int wave  = tid >> 5;        // N-tile id 0..15
  const int lane  = tid & 31;

  __shared__ __align__(16) unsigned short ldsA[16 * LDS_STRIDE];
  __shared__ float lattn[16];

  if (tid < 16) lattn[tid] = 0.0f;

  // --- cooperative load: 16 x 768 fp32 -> bf16 LDS (rows contiguous) ---
  const float* src = zone + ((size_t)b * Z_ + (size_t)ztile * 16) * D_;
  #pragma unroll
  for (int i = 0; i < 6; ++i) {
    const int f   = tid + i * 512;         // float4 index, 0..3071
    const int row = f / (D_ / 4);
    const int col = (f % (D_ / 4)) * 4;
    const float4 v4 = ((const float4*)src)[f];
    ushort4_t p;
    p[0] = f2bf(v4.x); p[1] = f2bf(v4.y); p[2] = f2bf(v4.z); p[3] = f2bf(v4.w);
    *(ushort4_t*)&ldsA[row * LDS_STRIDE + col] = p;
  }
  __syncthreads();

  // --- per-wave WMMA over K ---
  const int n    = lane & 15;        // N within tile (B/C fragment: lane%16 = N)
  const int half = lane >> 4;
  const int row  = n;                // A fragment: M = lane%16
  const int k0   = half ? 8 : 0;     // A fragment K sub-offsets per lane half
  const int k1   = half ? 24 : 16;

  // fold the 1/sqrt(FEATURE_DIM) scale into t (reduction is linear in t)
  const float tv  = tvec[(size_t)b * DO_ + wave * 16 + n] * INV_SQRT_FD;
  const float bzv = bz[wave * 16 + n];

  v8f c = {0.0f, 0.0f, 0.0f, 0.0f, 0.0f, 0.0f, 0.0f, 0.0f};

  #pragma unroll 6
  for (int kt = 0; kt < KT_; ++kt) {
    FragBF a, bb;
    a.h[0] = *(const ushort8*)&ldsA[row * LDS_STRIDE + kt * 32 + k0];
    a.h[1] = *(const ushort8*)&ldsA[row * LDS_STRIDE + kt * 32 + k1];
    const ushort8* bp =
        (const ushort8*)wz_bf + ((size_t)((kt * NT_ + wave) * 32 + lane)) * 2;
    bb.h[0] = bp[0];
    bb.h[1] = bp[1];
    c = __builtin_amdgcn_wmma_f32_16x16x32_bf16(
        /*neg_a=*/false, a.v, /*neg_b=*/false, bb.v,
        /*c_mod=*/(short)0, c, /*reuse_a=*/false, /*reuse_b=*/false);
  }

  // --- epilogue: tanh, scale by t, reduce over N (lanes within each half) ---
  // C layout: VGPR r, lane l -> M = r + 8*(l>>4), N = l&15
  #pragma unroll
  for (int r = 0; r < 8; ++r) {
    float e = fast_tanh(c[r] + bzv) * tv;
    e += __shfl_xor(e, 1);
    e += __shfl_xor(e, 2);
    e += __shfl_xor(e, 4);
    e += __shfl_xor(e, 8);           // full sum within each 16-lane half
    if (n == 0) atomicAdd(&lattn[r + 8 * half], e);
  }
  __syncthreads();

  if (tid < 16)
    attn[(size_t)b * Z_ + (size_t)ztile * 16 + tid] = lattn[tid];
}

// ---------------------------------------------------------------------------
// Kernel 4: masked softmax over Z per batch row
// ---------------------------------------------------------------------------
__global__ __launch_bounds__(256) void softmax_kernel(
    const float* __restrict__ attn, const int* __restrict__ mask,
    float* __restrict__ out)
{
  const int b = blockIdx.x, tid = threadIdx.x;
  __shared__ float red[256];
  const float* arow = attn + (size_t)b * Z_;
  const int*   mrow = mask + (size_t)b * Z_;

  float vals[Z_ / 256];
  float mx = -INFINITY;
  #pragma unroll
  for (int i = 0; i < Z_ / 256; ++i) {
    const int z = tid + i * 256;
    const float a = mrow[z] ? -INFINITY : arow[z];
    vals[i] = a;
    mx = fmaxf(mx, a);
  }
  red[tid] = mx;
  __syncthreads();
  for (int s = 128; s > 0; s >>= 1) {
    if (tid < s) red[tid] = fmaxf(red[tid], red[tid + s]);
    __syncthreads();
  }
  mx = red[0];
  __syncthreads();

  float sum = 0.0f;
  #pragma unroll
  for (int i = 0; i < Z_ / 256; ++i) {
    const float e = __expf(vals[i] - mx);   // exp(-inf - mx) = 0 for masked
    vals[i] = e;
    sum += e;
  }
  red[tid] = sum;
  __syncthreads();
  for (int s = 128; s > 0; s >>= 1) {
    if (tid < s) red[tid] += red[tid + s];
    __syncthreads();
  }
  const float inv = 1.0f / red[0];
  #pragma unroll
  for (int i = 0; i < Z_ / 256; ++i)
    out[(size_t)b * Z_ + tid + i * 256] = vals[i] * inv;
}

// ---------------------------------------------------------------------------
// Launcher
// ---------------------------------------------------------------------------
extern "C" void kernel_launch(void* const* d_in, const int* in_sizes, int n_in,
                              void* d_out, int out_size, void* d_ws, size_t ws_size,
                              hipStream_t stream) {
  const float* txt  = (const float*)d_in[0];   // [B, L, D]
  const float* zone = (const float*)d_in[1];   // [B, Z, D]
  const int*   mask = (const int*)d_in[2];     // [B, Z]
  const float* Wt   = (const float*)d_in[3];   // [D, DO]
  const float* bt   = (const float*)d_in[4];   // [DO]
  const float* Wz   = (const float*)d_in[5];   // [D, DO]
  const float* bz   = (const float*)d_in[6];   // [DO]
  float* out = (float*)d_out;                  // [B, Z]

  // workspace layout
  char* wsb = (char*)d_ws;
  float*          t_ws    = (float*)wsb;                         // 32 KB
  unsigned short* wz_ws   = (unsigned short*)(wsb + 32768);      // 384 KB
  float*          attn_ws = (float*)(wsb + 32768 + 393216);      // 512 KB

  prep_t_kernel<<<B_, 256, 0, stream>>>(txt, Wt, bt, t_ws);
  pack_wz_kernel<<<(KT_ * NT_ * 32 * 16) / 256, 256, 0, stream>>>(Wz, wz_ws);
  zone_attn_kernel<<<dim3(Z_ / 16, B_), 512, 0, stream>>>(zone, wz_ws, t_ws, bz, attn_ws);
  softmax_kernel<<<B_, 256, 0, stream>>>(attn_ws, mask, out);
}
